// RLogic_18365280158379
// MI455X (gfx1250) — compile-verified, matrix operand order checked by reading, not run on hardware
//
#include <hip/hip_runtime.h>

// ---------------- types ----------------
typedef __bf16 bf16_t;
typedef __attribute__((ext_vector_type(16))) __bf16 v16bf;
typedef __attribute__((ext_vector_type(8)))  __bf16 v8bf;
typedef __attribute__((ext_vector_type(8)))  float  v8f;

typedef int v4i __attribute__((vector_size(16)));
typedef v4i __attribute__((address_space(1)))* g4_ptr;   // global int4*
typedef v4i __attribute__((address_space(3)))* l4_ptr;   // LDS int4*

#if defined(__has_builtin)
#if __has_builtin(__builtin_amdgcn_global_load_async_to_lds_b128)
#define HAVE_ASYNC_LDS 1
#endif
#endif

__device__ __forceinline__ bf16_t f2bf(float f) {
    unsigned u = __builtin_bit_cast(unsigned, f);
    unsigned r = (u + 0x7FFFu + ((u >> 16) & 1u)) >> 16;
    unsigned short s = (unsigned short)r;
    return __builtin_bit_cast(bf16_t, s);
}
__device__ __forceinline__ float sigmoidf_(float x) { return 1.0f / (1.0f + __expf(-x)); }

// 16B global -> LDS copy (async on CDNA5, tracked by ASYNCcnt)
__device__ __forceinline__ void cp16(const bf16_t* g, bf16_t* l) {
#ifdef HAVE_ASYNC_LDS
    g4_ptr gp = (g4_ptr)(uintptr_t)g;
    l4_ptr lp = (l4_ptr)(unsigned)(uintptr_t)l;
    __builtin_amdgcn_global_load_async_to_lds_b128(gp, lp, 0, 0);
#else
    *(uint4*)l = *(const uint4*)g;
#endif
}
__device__ __forceinline__ void wait_async_lds() {
#ifdef HAVE_ASYNC_LDS
#if __has_builtin(__builtin_amdgcn_s_wait_asynccnt)
    __builtin_amdgcn_s_wait_asynccnt(0);
#else
    asm volatile("s_wait_asynccnt 0x0" ::: "memory");
#endif
#endif
}

// ---------------- generic WMMA GEMM ----------------
// C[M x N] = A[M x K](bf16) * W[N x K]^T(bf16)  (+ A2 * W2^T)  (+ bias + bias2)
// Block tile 64x256, 8 waves (2x4); each wave: 32x64 = 2x4 wmma 16x16x32 tiles.
// Requirements: M % 64 == 0, N % 256 == 0 (buffers padded), K % 32 == 0.
// Per-element store guard n < Nout handles logical N (e.g. 10001).
#define BLK_M 64
#define BLK_N 256
#define LDSP  40   // LDS row pitch in halves (80B, 16B-aligned, conflict padding)

__global__ void __launch_bounds__(256)
wmma_gemm(const bf16_t* __restrict__ A,  int lda,  const bf16_t* __restrict__ Bw,  int ldb,  int K,
          const bf16_t* __restrict__ A2, int lda2, const bf16_t* __restrict__ B2,  int ldb2, int K2,
          const float*  __restrict__ bias, const float* __restrict__ bias2,
          float* __restrict__ Cf, int ldc, bf16_t* __restrict__ Cbf, int ldcb,
          int M, int N, int Nout, int relu)
{
    __shared__ bf16_t As[2][BLK_M * LDSP];
    __shared__ bf16_t Bs[2][BLK_N * LDSP];

    const int tid  = threadIdx.x;
    const int lane = tid & 31;
    const int wave = tid >> 5;
    const int wm   = wave >> 2;       // 0..1  (32-row slab)
    const int wn   = wave & 3;        // 0..3  (64-col slab)
    const int blockM = blockIdx.y * BLK_M;
    const int blockN = blockIdx.x * BLK_N;

    // staging map (unconditional; buffers are padded)
    const int arow = tid >> 2, ac = tid & 3;   // A: 64 rows x 4 chunks of 16B
    const int brow = tid;                      // B: 256 rows x 4 chunks of 16B each

    v8f acc[2][4] = {};

    for (int pass = 0; pass < 2; ++pass) {
        const bf16_t* Ap = pass ? A2 : A;
        const bf16_t* Bp = pass ? B2 : Bw;
        const int ldap = pass ? lda2 : lda;
        const int ldbp = pass ? ldb2 : ldb;
        const int Kp   = pass ? K2   : K;
        if (Ap == nullptr || Bp == nullptr || Kp <= 0) continue;

        const bf16_t* Arow = Ap + (size_t)(blockM + arow) * ldap;
        const bf16_t* Brow = Bp + (size_t)(blockN + brow) * ldbp;

        // prologue: stage k-tile 0 into buffer 0
        cp16(Arow + 0 + ac * 8, &As[0][arow * LDSP + ac * 8]);
        #pragma unroll
        for (int c = 0; c < 4; ++c)
            cp16(Brow + 0 + c * 8, &Bs[0][brow * LDSP + c * 8]);
        wait_async_lds();
        __syncthreads();

        int cur = 0;
        for (int kk = 0; kk < Kp; kk += 32) {
            // issue async stage of next k-tile into the other buffer
            if (kk + 32 < Kp) {
                const int nxt = cur ^ 1;
                cp16(Arow + kk + 32 + ac * 8, &As[nxt][arow * LDSP + ac * 8]);
                #pragma unroll
                for (int c = 0; c < 4; ++c)
                    cp16(Brow + kk + 32 + c * 8, &Bs[nxt][brow * LDSP + c * 8]);
                __builtin_prefetch(Arow + kk + 64 + ac * 8, 0, 1);
            }

            // ---- fragments (CDNA5 16-bit A/B layout):
            // lanes 0-15: row = rsel, K {0..7,16..23}; lanes 16-31: K {8..15,24..31}
            const int sub  = (lane >> 4) * 8;
            const int rsel = lane & 15;
            v16bf afrag[2], bfrag[4];
            #pragma unroll
            for (int mt = 0; mt < 2; ++mt) {
                const int r = wm * 32 + mt * 16 + rsel;
                v8bf lo = *(const v8bf*)(&As[cur][r * LDSP + sub]);
                v8bf hi = *(const v8bf*)(&As[cur][r * LDSP + 16 + sub]);
                afrag[mt] = __builtin_shufflevector(lo, hi, 0,1,2,3,4,5,6,7,8,9,10,11,12,13,14,15);
            }
            #pragma unroll
            for (int nt = 0; nt < 4; ++nt) {
                const int r = wn * 64 + nt * 16 + rsel;
                v8bf lo = *(const v8bf*)(&Bs[cur][r * LDSP + sub]);
                v8bf hi = *(const v8bf*)(&Bs[cur][r * LDSP + 16 + sub]);
                bfrag[nt] = __builtin_shufflevector(lo, hi, 0,1,2,3,4,5,6,7,8,9,10,11,12,13,14,15);
            }
            #pragma unroll
            for (int mt = 0; mt < 2; ++mt)
                #pragma unroll
                for (int nt = 0; nt < 4; ++nt)
                    acc[mt][nt] = __builtin_amdgcn_wmma_f32_16x16x32_bf16(
                        false, afrag[mt], false, bfrag[nt],
                        (short)0, acc[mt][nt], false, false);

            wait_async_lds();
            __syncthreads();
            cur ^= 1;
        }
    }

    // ---- epilogue: VGPR r -> M = r (+8 for lanes 16-31), lane&15 -> N
    const int laneN = lane & 15;
    const int moff  = (lane >> 4) * 8;
    #pragma unroll
    for (int mt = 0; mt < 2; ++mt) {
        #pragma unroll
        for (int nt = 0; nt < 4; ++nt) {
            const int n = blockN + wn * 64 + nt * 16 + laneN;
            float bv = 0.0f;
            if (bias)  bv += bias[n];
            if (bias2) bv += bias2[n];
            #pragma unroll
            for (int r = 0; r < 8; ++r) {
                const int m = blockM + wm * 32 + mt * 16 + r + moff;
                if (m < M && n < Nout) {
                    float v = acc[mt][nt][r] + bv;
                    if (relu && v < 0.0f) v = 0.0f;
                    if (Cf)  Cf[(size_t)m * ldc + n] = v;
                    if (Cbf) Cbf[(size_t)m * ldcb + n] = f2bf(v);
                }
            }
        }
    }
}

// ---------------- elementwise kernels ----------------
__global__ void conv_bf(const float* __restrict__ s, bf16_t* __restrict__ d, int n) {
    int i = blockIdx.x * 256 + threadIdx.x;
    if (i < n) d[i] = f2bf(s[i]);
}
__global__ void conv_bf_pad(const float* __restrict__ s, bf16_t* __restrict__ d, int nsrc, int ntot) {
    int i = blockIdx.x * 256 + threadIdx.x;
    if (i < ntot) d[i] = (i < nsrc) ? f2bf(s[i]) : f2bf(0.0f);
}
__global__ void pad_bias(const float* __restrict__ s, float* __restrict__ d, int nsrc, int ntot) {
    int i = blockIdx.x * 256 + threadIdx.x;
    if (i < ntot) d[i] = (i < nsrc) ? s[i] : 0.0f;
}
// emb_w (R x E) -> embT (E x NP) bf16, zero-padded K
__global__ void build_embT(const float* __restrict__ emb_w, bf16_t* __restrict__ dst,
                           int E, int NP, int R) {
    int i = blockIdx.x * 256 + threadIdx.x;
    if (i >= E * NP) return;
    int e = i / NP, r = i - e * NP;
    dst[i] = (r < R) ? f2bf(emb_w[(size_t)r * E + e]) : f2bf(0.0f);
}
__global__ void gather_x(const int* __restrict__ bodys, const float* __restrict__ emb_w,
                         bf16_t* __restrict__ xbf, int total) {
    int i = blockIdx.x * 256 + threadIdx.x;
    if (i >= total) return;
    int e = i & 255, bl = i >> 8;
    xbf[i] = f2bf(emb_w[(size_t)bodys[bl] * 256 + e]);
}
__global__ void init_hc(bf16_t* __restrict__ h, float* __restrict__ c, int n) {
    int i = blockIdx.x * 256 + threadIdx.x;
    if (i < n) { h[i] = f2bf(0.0f); c[i] = 0.0f; }
}
__global__ void lstm_cell(const float* __restrict__ G, float* __restrict__ cst,
                          bf16_t* __restrict__ hbf, float* __restrict__ body_hid, int t) {
    int i = blockIdx.x * 256 + threadIdx.x;   // over 1024*256
    int b = i >> 8, e = i & 255;
    const float* g = G + (size_t)b * 1024;
    float gi = g[e], gf = g[e + 256], gg = g[e + 512], go = g[e + 768];
    float c = sigmoidf_(gf) * cst[i] + sigmoidf_(gi) * tanhf(gg);
    float h = sigmoidf_(go) * tanhf(c);
    cst[i] = c;
    hbf[i] = f2bf(h);
    body_hid[((size_t)b * 8 + t) * 256 + e] = h;
}
__global__ void __launch_bounds__(256)
softmax_row(const float* __restrict__ prob, int ldp, int ncols,
            bf16_t* __restrict__ psf, int npad, float* __restrict__ sR, int colR) {
    __shared__ float red[256];
    const int b = blockIdx.x;
    const float* row = prob + (size_t)b * ldp;
    float mx = -3.0e38f;
    for (int j = threadIdx.x; j < ncols; j += 256) mx = fmaxf(mx, row[j]);
    red[threadIdx.x] = mx; __syncthreads();
    for (int s = 128; s > 0; s >>= 1) {
        if ((int)threadIdx.x < s) red[threadIdx.x] = fmaxf(red[threadIdx.x], red[threadIdx.x + s]);
        __syncthreads();
    }
    mx = red[0]; __syncthreads();
    float sum = 0.0f;
    for (int j = threadIdx.x; j < ncols; j += 256) sum += __expf(row[j] - mx);
    red[threadIdx.x] = sum; __syncthreads();
    for (int s = 128; s > 0; s >>= 1) {
        if ((int)threadIdx.x < s) red[threadIdx.x] += red[threadIdx.x + s];
        __syncthreads();
    }
    const float inv = 1.0f / red[0];
    bf16_t* orow = psf + (size_t)b * npad;
    for (int j = threadIdx.x; j < npad; j += 256) {
        float v = (j < ncols) ? __expf(row[j] - mx) * inv : 0.0f;
        orow[j] = f2bf(v);
        if (j == colR) sR[b] = v;
    }
}
// emb_concat builder: writes f32 output region + bf16 GEMM input
__global__ void build_concat(const int* __restrict__ bodys, const float* __restrict__ emb_w,
                             const float* __restrict__ emb1_t, const float* __restrict__ sR,
                             const float* __restrict__ body_hid, int it,
                             float* __restrict__ outc, bf16_t* __restrict__ catbf) {
    int i = blockIdx.x * 256 + threadIdx.x;   // over 1024*256
    int b = i >> 8, e = i & 255;
    float e1, sec;
    if (it == 0) {
        e1  = emb_w[(size_t)bodys[b * 8 + 0] * 256 + e];
        sec = emb_w[(size_t)bodys[b * 8 + 1] * 256 + e];
    } else {
        e1  = emb1_t[i] + sR[b] * body_hid[((size_t)b * 8 + it) * 256 + e];
        sec = emb_w[(size_t)(it + 1) * 256 + e];
    }
    outc[(size_t)b * 512 + e] = e1;
    outc[(size_t)b * 512 + 256 + e] = sec;
    catbf[(size_t)b * 512 + e] = f2bf(e1);
    catbf[(size_t)b * 512 + 256 + e] = f2bf(sec);
}

// ---------------- host launch ----------------
extern "C" void kernel_launch(void* const* d_in, const int* in_sizes, int n_in,
                              void* d_out, int out_size, void* d_ws, size_t ws_size,
                              hipStream_t stream)
{
    const int B = 1024, L = 8, E = 256, R = 10000;
    const int NP   = 10016;      // (R+1) padded to multiple of 32  (K of psf GEMM)
    const int NFC2 = 10240;      // (R+1) padded to multiple of BLK_N (N of fc2 GEMM)
    const int G4 = 4 * E;        // 1024

    const int*   bodys = (const int*)  d_in[0];
    const float* emb_w = (const float*)d_in[1];
    const float* w_ih  = (const float*)d_in[2];
    const float* w_hh  = (const float*)d_in[3];
    const float* b_ih  = (const float*)d_in[4];
    const float* b_hh  = (const float*)d_in[5];
    const float* fc1_w = (const float*)d_in[6];
    const float* fc1_b = (const float*)d_in[7];
    const float* fc2_w = (const float*)d_in[8];
    const float* fc2_b = (const float*)d_in[9];

    float* out_prob   = (float*)d_out;                 // (1024, 10001)
    float* out_concat = out_prob + (size_t)B * (R + 1);// (1024, 512)

    char* ws = (char*)d_ws;
    size_t off = 0;
    auto alloc = [&](size_t bytes) -> char* {
        char* p = ws + off;
        off = (off + bytes + 255) & ~(size_t)255;
        return p;
    };
    bf16_t* x_bf    = (bf16_t*)alloc((size_t)B * L * E * 2);
    bf16_t* h_bf    = (bf16_t*)alloc((size_t)B * E * 2);
    bf16_t* wih_bf  = (bf16_t*)alloc((size_t)G4 * E * 2);
    bf16_t* whh_bf  = (bf16_t*)alloc((size_t)G4 * E * 2);
    bf16_t* fc1w_bf = (bf16_t*)alloc((size_t)E * 2 * E * 2);
    bf16_t* fc2w_bf = (bf16_t*)alloc((size_t)NFC2 * E * 2);
    bf16_t* embT_bf = (bf16_t*)alloc((size_t)E * NP * 2);
    float*  fc2b_p  = (float*) alloc((size_t)NFC2 * 4);
    float*  Gbuf    = (float*) alloc((size_t)B * G4 * 4);
    float*  c_st    = (float*) alloc((size_t)B * E * 4);
    float*  bodyh   = (float*) alloc((size_t)B * L * E * 4);
    float*  emb1_t  = (float*) alloc((size_t)B * E * 4);
    bf16_t* cat_bf  = (bf16_t*)alloc((size_t)B * 2 * E * 2);
    bf16_t* act_bf  = (bf16_t*)alloc((size_t)B * E * 2);
    bf16_t* psf_bf  = (bf16_t*)alloc((size_t)B * NP * 2);
    float*  sR      = (float*) alloc((size_t)B * 4);

    // ---- weight conversion / padding
    conv_bf<<<dim3((G4 * E + 255) / 256), 256, 0, stream>>>(w_ih, wih_bf, G4 * E);
    conv_bf<<<dim3((G4 * E + 255) / 256), 256, 0, stream>>>(w_hh, whh_bf, G4 * E);
    conv_bf<<<dim3((E * 2 * E + 255) / 256), 256, 0, stream>>>(fc1_w, fc1w_bf, E * 2 * E);
    conv_bf_pad<<<dim3((NFC2 * E + 255) / 256), 256, 0, stream>>>(fc2_w, fc2w_bf, (R + 1) * E, NFC2 * E);
    pad_bias<<<dim3((NFC2 + 255) / 256), 256, 0, stream>>>(fc2_b, fc2b_p, R + 1, NFC2);
    build_embT<<<dim3((E * NP + 255) / 256), 256, 0, stream>>>(emb_w, embT_bf, E, NP, R);
    gather_x<<<dim3((B * L * E) / 256), 256, 0, stream>>>(bodys, emb_w, x_bf, B * L * E);
    init_hc<<<dim3((B * E) / 256), 256, 0, stream>>>(h_bf, c_st, B * E);

    // ---- LSTM: G = x_t @ Wih^T + h @ Whh^T + b_ih + b_hh, then cell update
    dim3 gGates(G4 / BLK_N, B / BLK_M);              // (4,16)
    for (int t = 0; t < L; ++t) {
        wmma_gemm<<<gGates, 256, 0, stream>>>(
            x_bf + (size_t)t * E, L * E, wih_bf, E, E,
            h_bf, E, whh_bf, E, E,
            b_ih, b_hh,
            Gbuf, G4, nullptr, 0,
            B, G4, G4, 0);
        lstm_cell<<<dim3((B * E) / 256), 256, 0, stream>>>(Gbuf, c_st, h_bf, bodyh, t);
    }

    // ---- decode loop
    dim3 gSmall(E / BLK_N ? E / BLK_N : 1, B / BLK_M);     // (1,16)
    dim3 gFc2(NFC2 / BLK_N, B / BLK_M);                    // (40,16)
    for (int it = 0; it < L - 1; ++it) {
        if (it > 0) {
            // emb1 = prob_sf @ emb_w  (A: B x NP, W: embT E x NP)
            wmma_gemm<<<gSmall, 256, 0, stream>>>(
                psf_bf, NP, embT_bf, NP, NP,
                nullptr, 0, nullptr, 0, 0,
                nullptr, nullptr,
                emb1_t, E, nullptr, 0,
                B, E, E, 0);
        }
        build_concat<<<dim3((B * E) / 256), 256, 0, stream>>>(
            bodys, emb_w, emb1_t, sR, bodyh, it, out_concat, cat_bf);
        // fc1 + relu -> bf16 activations
        wmma_gemm<<<gSmall, 256, 0, stream>>>(
            cat_bf, 2 * E, fc1w_bf, 2 * E, 2 * E,
            nullptr, 0, nullptr, 0, 0,
            fc1_b, nullptr,
            nullptr, 0, act_bf, E,
            B, E, E, 1);
        // fc2 -> prob (f32, directly into d_out, stores guarded to 10001)
        wmma_gemm<<<gFc2, 256, 0, stream>>>(
            act_bf, E, fc2w_bf, E, E,
            nullptr, 0, nullptr, 0, 0,
            fc2b_p, nullptr,
            out_prob, R + 1, nullptr, 0,
            B, NFC2, R + 1, 0);
        // softmax -> padded bf16 prob_sf + scalar column R
        softmax_row<<<dim3(B), 256, 0, stream>>>(out_prob, R + 1, R + 1, psf_bf, NP, sR, R);
    }
}
